// CrossAxisAttention_40647570489983
// MI455X (gfx1250) — matrix operand order repaired and checked by reading, not compiled, not run on hardware
//
#include <hip/hip_runtime.h>
#include <cstdint>
#include <cstddef>

// ---------------------------------------------------------------------------
// CrossAxisAttention for MI455X (gfx1250): bf16 WMMA pipeline
//   x[8,256,56,56] --(qkv gemm)--> q,k,v --(dw3x3 on v)--> stripe flash-attn
//   (4 heads H-stripes, 4 heads W-stripes, sw=7, L=392, d=32) --(proj gemm)--> out
// Matrix math via v_wmma_f32_16x16x32_bf16; K staging via TDM tensor_load_to_lds.
// ---------------------------------------------------------------------------

typedef unsigned short u16;
typedef __attribute__((ext_vector_type(16))) __bf16 v16bf;
typedef __attribute__((ext_vector_type(8)))  float  v8f;

struct U16x16 { uint4 a, b; };   // 32 bytes == one 16x(bf16) WMMA operand

__device__ __forceinline__ v16bf as_frag(U16x16 u) {
    return __builtin_bit_cast(v16bf, u);
}

__device__ __forceinline__ u16 f2bf(float f) {
    unsigned u = __builtin_bit_cast(unsigned, f);
    unsigned r = u + 0x7FFFu + ((u >> 16) & 1u);   // round-to-nearest-even
    return (u16)(r >> 16);
}

#define NTOK   25088          // 8 * 56 * 56
#define HWPIX  3136           // 56*56
#define CDIM   256
#define SCALEQ 0.17677669529663687f   // 32^-0.5

// ---- Tensor Data Mover availability (device pass only; host pass falls back)
#if defined(__HIP_DEVICE_COMPILE__) && \
    __has_builtin(__builtin_amdgcn_tensor_load_to_lds) && \
    __has_builtin(__builtin_amdgcn_s_wait_tensorcnt)
#define USE_TDM 1
#else
#define USE_TDM 0
#endif

#if USE_TDM
typedef __attribute__((ext_vector_type(4))) unsigned int tdm_u32x4;
typedef __attribute__((ext_vector_type(8))) int          tdm_i32x8;
typedef __attribute__((ext_vector_type(4))) int          tdm_i32x4;
struct TdmG0 { unsigned w[4]; };
struct TdmG1 { unsigned w[8]; };
struct TdmG2 { unsigned w[4]; };

// set bits [lo, lo+nb) of a little-endian dword array (uniform/scalar use)
__device__ __forceinline__ void dput(unsigned* w, int lo, int nb,
                                     unsigned long long v) {
    if (nb < 64) v &= ((1ull << nb) - 1ull);
    int word = lo >> 5, sh = lo & 31;
    w[word] |= (unsigned)(v << sh);
    if (sh + nb > 32) w[word + 1] |= (unsigned)(v >> (32 - sh));
    if (sh + nb > 64) w[word + 2] |= (unsigned)(v >> (64 - sh));
}
#endif

// A fragment (16x32 bf16, M x K): lane L -> row M=L%16.
//   elems 0..15 : k = (L>=16?16:0) + i       (32B contiguous)
__device__ __forceinline__ v16bf load_afrag(const u16* base, int ldk, int lane) {
    const u16* p = base + ((lane >> 4) << 4);
    U16x16 f;
    f.a = *(const uint4*)(p);
    f.b = *(const uint4*)(p + 8);
    (void)ldk;
    return as_frag(f);
}

// B fragment (32x16 bf16, K x N) from a row-major [N][>=32] matrix where
// B[k,n] = M[n][k]: lane L -> col N=L%16, elems i -> k = (L>=16?16:0)+i.
__device__ __forceinline__ v16bf load_bfrag_rowmajor(const u16* mat, int ld,
                                                     int n0, int kcol, int lane) {
    const u16* p = mat + (size_t)(n0 + (lane & 15)) * ld + kcol + ((lane >> 4) << 4);
    U16x16 f;
    f.a = *(const uint4*)(p);
    f.b = *(const uint4*)(p + 8);
    return as_frag(f);
}

// ---------------------------------------------------------------------------
// Stage 0: fp32 -> bf16 converters
// ---------------------------------------------------------------------------
__global__ __launch_bounds__(256) void cvt_w_kernel(const float* __restrict__ src,
                                                    u16* __restrict__ dst, int n) {
    for (int i = blockIdx.x * 256 + threadIdx.x; i < n; i += gridDim.x * 256)
        dst[i] = f2bf(src[i]);
}

// x[B,C,H,W] fp32 -> Xt[token][C] bf16  (token = b*3136 + h*56 + w)
__global__ __launch_bounds__(256) void cvt_x_kernel(const float* __restrict__ x,
                                                    u16* __restrict__ Xt) {
    int idx = blockIdx.x * 256 + threadIdx.x;     // [0, 25088*256)
    int c   = idx & 255;
    int tok = idx >> 8;
    int b   = tok / HWPIX;
    int p   = tok - b * HWPIX;
    Xt[idx] = f2bf(x[(size_t)b * (CDIM * HWPIX) + (size_t)c * HWPIX + p]);
}

// ---------------------------------------------------------------------------
// Stage 1: QKV GEMM  D[25088,768] = Xt[25088,256] * Wqkv^T + b
//   grid (196, 48), block 256 (8 waves); wave -> 16x16 tile, 8 K-steps.
// ---------------------------------------------------------------------------
__global__ __launch_bounds__(256) void gemm_qkv_kernel(const u16* __restrict__ Xt,
                                                       const u16* __restrict__ W,
                                                       const float* __restrict__ bias,
                                                       u16* __restrict__ Qb,
                                                       u16* __restrict__ Kb,
                                                       float* __restrict__ Vf) {
    const int wave = threadIdx.x >> 5, lane = threadIdx.x & 31;
    const int nlo = lane & 15, half = lane >> 4;
    const int m0 = blockIdx.x * 128 + wave * 16;
    const int nt = blockIdx.y;
    const int n0 = nt * 16;

    v8f acc = {};
    const u16* arow = Xt + (size_t)(m0 + nlo) * CDIM;
    __builtin_prefetch(W + (size_t)(n0 + nlo) * CDIM, 0, 0);   // global_prefetch_b8
#pragma unroll
    for (int kk = 0; kk < 256; kk += 32) {
        v16bf a = load_afrag(arow + kk, CDIM, lane);
        v16bf b = load_bfrag_rowmajor(W, CDIM, n0, kk, lane);
        acc = __builtin_amdgcn_wmma_f32_16x16x32_bf16(false, a, false, b,
                                                      (short)0, acc, false, false);
    }
    const int col = n0 + nlo;
    const float bv = bias[col];
#pragma unroll
    for (int j = 0; j < 8; ++j) {
        int row = m0 + j + half * 8;
        float r = acc[j] + bv;
        if (nt < 16)        Qb[(size_t)row * CDIM + col]         = f2bf(r * SCALEQ);
        else if (nt < 32)   Kb[(size_t)row * CDIM + (col - 256)] = f2bf(r);
        else                Vf[(size_t)row * CDIM + (col - 512)] = r;
    }
}

// ---------------------------------------------------------------------------
// Stage 2: depthwise 3x3 (SAME) residual:  vfin = bf16(v + dw3x3(v) + b_dw)
// ---------------------------------------------------------------------------
__global__ __launch_bounds__(256) void dwconv_kernel(const float* __restrict__ V,
                                                     const float* __restrict__ wdw,
                                                     const float* __restrict__ bdw,
                                                     u16* __restrict__ Vout) {
    int idx = blockIdx.x * 256 + threadIdx.x;     // [0, 25088*256)
    int c   = idx & 255;
    int tok = idx >> 8;
    int b   = tok / HWPIX;
    int p   = tok - b * HWPIX;
    int h = p / 56, w = p - h * 56;
    float s = 0.f;
#pragma unroll
    for (int dy = -1; dy <= 1; ++dy) {
#pragma unroll
        for (int dx = -1; dx <= 1; ++dx) {
            int hh = h + dy, ww = w + dx;
            if (hh >= 0 && hh < 56 && ww >= 0 && ww < 56) {
                float wv = wdw[c * 9 + (dy + 1) * 3 + (dx + 1)];
                s += wv * V[((size_t)b * HWPIX + hh * 56 + ww) * CDIM + c];
            }
        }
    }
    Vout[idx] = f2bf(V[idx] + s + bdw[c]);
}

// ---------------------------------------------------------------------------
// Stage 3: stripe flash-attention.
//   One workgroup per (b, head, stripe): 512 blocks, 256 threads (8 waves).
//   L=392 tokens (padded 416 = 13 chunks of 32 keys), d=32.
//   K tile staged via TDM (tensor_load_to_lds); V staged transposed by threads.
// ---------------------------------------------------------------------------
__global__ __launch_bounds__(256) void stripe_attn_kernel(const u16* __restrict__ Qb,
                                                          const u16* __restrict__ Kb,
                                                          const u16* __restrict__ Vb,
                                                          u16* __restrict__ Ob) {
    __shared__ __align__(16) u16 Ks[416 * 32];      // K rows, row stride 32
    __shared__ __align__(16) u16 Vt[32 * 416];      // V transposed: [d][key]
    __shared__ __align__(16) u16 Pb[8 * 16 * 32];   // per-wave P staging

    const int sidx = blockIdx.x;          // 512
    const int b    = sidx >> 6;
    const int rr   = sidx & 63;
    const int head = rr >> 3;
    const int sn   = rr & 7;              // stripe index
    const int dir  = head >> 2;           // 0: H-stripes, 1: W-stripes
    const int ho   = head * 32;
    const int base = b * HWPIX;

#if USE_TDM
    // ---- K tile via Tensor Data Mover -----------------------------------
    // dir0: 2D tile 32ch x 392 rows, row stride 256 elems (2B each).
    // dir1: 3D tile X=32ch, Y=56 (h, stride 56*256), Z=7 (s, stride 256);
    //       TDM Z/Y/X fill order == stripe row order idx = s*56 + h.
    if (threadIdx.x < 32) {               // wave 0 issues + waits (TENSORcnt)
        TdmG0 g0 = {}; TdmG1 g1 = {}; TdmG2 g2 = {}; TdmG2 g3 = {};
        const int gtok0 = (dir == 0) ? (base + sn * 392) : (base + sn * 7);
        unsigned long long gaddr =
            (unsigned long long)(uintptr_t)(Kb + (size_t)gtok0 * CDIM + ho);
        unsigned ldsoff = (unsigned)(uintptr_t)(void*)Ks;   // LDS byte offset

        dput(g0.w, 0, 2, 1);                 // count = 1 (valid descriptor)
        dput(g0.w, 32, 32, ldsoff);          // lds_addr
        dput(g0.w, 64, 57, gaddr);           // global_addr (bytes)
        dput(g0.w, 126, 2, 2);               // type = 2 ("image")

        dput(g1.w, 16, 2, 1);                // data_size = 2 bytes
        dput(g1.w, 48, 32, 32);              // tensor_dim0 = 32
        dput(g1.w, 80, 32, dir == 0 ? 392 : 56);   // tensor_dim1
        dput(g1.w, 112, 16, 32);             // tile_dim0 = 32
        dput(g1.w, 128, 16, dir == 0 ? 392 : 56);  // tile_dim1
        dput(g1.w, 144, 16, dir == 0 ? 0 : 7);     // tile_dim2
        dput(g1.w, 160, 48, dir == 0 ? 256 : 14336); // tensor_dim0_stride
        dput(g1.w, 208, 48, dir == 0 ? 0 : 256);     // tensor_dim1_stride

        if (dir == 1) g2.w[0] = 7;           // tensor_dim2

        tdm_u32x4 vg0 = __builtin_bit_cast(tdm_u32x4, g0);
        tdm_i32x8 vg1 = __builtin_bit_cast(tdm_i32x8, g1);
        tdm_i32x4 vg2 = __builtin_bit_cast(tdm_i32x4, g2);
        tdm_i32x4 vg3 = __builtin_bit_cast(tdm_i32x4, g3);
#if defined(__clang_major__) && (__clang_major__ >= 23)
        tdm_i32x8 vzero = {};
        __builtin_amdgcn_tensor_load_to_lds(vg0, vg1, vg2, vg3, vzero, 0);
#else
        __builtin_amdgcn_tensor_load_to_lds(vg0, vg1, vg2, vg3, 0);
#endif
        __builtin_amdgcn_s_wait_tensorcnt((short)0);
    }
    // ---- V (transposed) by threads; zero K padding rows -----------------
    for (int row = threadIdx.x; row < 416; row += 256) {
        uint4 vq[4] = {};
        if (row < 392) {
            int tok = (dir == 0) ? (base + sn * 392 + row)
                                 : (base + (row % 56) * 56 + sn * 7 + row / 56);
            const uint4* vp = (const uint4*)(Vb + (size_t)tok * CDIM + ho);
            vq[0] = vp[0]; vq[1] = vp[1]; vq[2] = vp[2]; vq[3] = vp[3];
        } else {
            uint4 z = {};
            uint4* kd = (uint4*)(Ks + row * 32);
            kd[0] = z; kd[1] = z; kd[2] = z; kd[3] = z;
        }
        union { uint4 q[4]; u16 s[32]; } uv;
        uv.q[0] = vq[0]; uv.q[1] = vq[1]; uv.q[2] = vq[2]; uv.q[3] = vq[3];
#pragma unroll
        for (int c = 0; c < 32; ++c) Vt[c * 416 + row] = uv.s[c];
    }
#else
    // ---- fallback: both K and V staged with plain loads -----------------
    for (int row = threadIdx.x; row < 416; row += 256) {
        uint4 kq[4] = {};
        uint4 vq[4] = {};
        if (row < 392) {
            int tok = (dir == 0) ? (base + sn * 392 + row)
                                 : (base + (row % 56) * 56 + sn * 7 + row / 56);
            const uint4* kp = (const uint4*)(Kb + (size_t)tok * CDIM + ho);
            const uint4* vp = (const uint4*)(Vb + (size_t)tok * CDIM + ho);
            kq[0] = kp[0]; kq[1] = kp[1]; kq[2] = kp[2]; kq[3] = kp[3];
            vq[0] = vp[0]; vq[1] = vp[1]; vq[2] = vp[2]; vq[3] = vp[3];
        }
        uint4* kd = (uint4*)(Ks + row * 32);
        kd[0] = kq[0]; kd[1] = kq[1]; kd[2] = kq[2]; kd[3] = kq[3];
        union { uint4 q[4]; u16 s[32]; } uv;
        uv.q[0] = vq[0]; uv.q[1] = vq[1]; uv.q[2] = vq[2]; uv.q[3] = vq[3];
#pragma unroll
        for (int c = 0; c < 32; ++c) Vt[c * 416 + row] = uv.s[c];
    }
#endif
    __syncthreads();

    const int wave = threadIdx.x >> 5, lane = threadIdx.x & 31;
    const int nlo = lane & 15, half = lane >> 4;
    u16* Pw = Pb + wave * 512;

    for (int t = wave; t < 25; t += 8) {          // 25 query tiles of 16
        const int q0 = t * 16;
        // Q A-fragment: lane row = q0 + nlo (clamped for the padded tail)
        int qi = q0 + nlo; if (qi > 391) qi = 391;
        int qtok = (dir == 0) ? (base + sn * 392 + qi)
                              : (base + (qi % 56) * 56 + sn * 7 + qi / 56);
        v16bf aQ = load_afrag(Qb + (size_t)qtok * CDIM + ho, CDIM, lane);

        v8f acc0 = {}, acc1 = {};                 // O cols 0-15 / 16-31
        float mrow[8], lrow[8];
#pragma unroll
        for (int j = 0; j < 8; ++j) { mrow[j] = -1e30f; lrow[j] = 0.f; }

        for (int kc = 0; kc < 13; ++kc) {
            const int k0 = kc * 32;
            // S = Q * K^T for 32 keys (two 16-col tiles)
            v16bf bK0 = load_bfrag_rowmajor(Ks, 32, k0,      0, lane);
            v16bf bK1 = load_bfrag_rowmajor(Ks, 32, k0 + 16, 0, lane);
            v8f z = {};
            v8f s0 = __builtin_amdgcn_wmma_f32_16x16x32_bf16(false, aQ, false, bK0,
                                                             (short)0, z, false, false);
            v8f s1 = __builtin_amdgcn_wmma_f32_16x16x32_bf16(false, aQ, false, bK1,
                                                             (short)0, z, false, false);
            const bool v0 = (k0 + nlo) < 392;
            const bool v1 = (k0 + 16 + nlo) < 392;
            // ---- online softmax over this 32-key chunk ----
#pragma unroll
            for (int j = 0; j < 8; ++j) {
                float a0 = v0 ? s0[j] : -1e30f;
                float a1 = v1 ? s1[j] : -1e30f;
                float mx = fmaxf(a0, a1);
                for (int off = 1; off < 16; off <<= 1)
                    mx = fmaxf(mx, __shfl_xor(mx, off, 16));
                float mnew = fmaxf(mrow[j], mx);
                float sc = __expf(mrow[j] - mnew);
                float p0 = v0 ? __expf(a0 - mnew) : 0.f;
                float p1 = v1 ? __expf(a1 - mnew) : 0.f;
                float rs = p0 + p1;
                for (int off = 1; off < 16; off <<= 1)
                    rs += __shfl_xor(rs, off, 16);
                lrow[j] = lrow[j] * sc + rs;
                mrow[j] = mnew;
                acc0[j] *= sc;
                acc1[j] *= sc;
                // P -> LDS in [M][32] layout (C-frag element j -> row j+half*8)
                Pw[(j + half * 8) * 32 + nlo]      = f2bf(p0);
                Pw[(j + half * 8) * 32 + 16 + nlo] = f2bf(p1);
            }
            // O += P * V : P as A-fragment, V columns from transposed LDS
            v16bf aP = load_afrag(Pw + nlo * 32, 32, lane);
            {
                const u16* p0 = Vt + nlo * 416 + k0 + half * 16;
                const u16* p1 = Vt + (16 + nlo) * 416 + k0 + half * 16;
                U16x16 f0, f1;
                f0.a = *(const uint4*)(p0); f0.b = *(const uint4*)(p0 + 8);
                f1.a = *(const uint4*)(p1); f1.b = *(const uint4*)(p1 + 8);
                acc0 = __builtin_amdgcn_wmma_f32_16x16x32_bf16(false, aP, false, as_frag(f0),
                                                               (short)0, acc0, false, false);
                acc1 = __builtin_amdgcn_wmma_f32_16x16x32_bf16(false, aP, false, as_frag(f1),
                                                               (short)0, acc1, false, false);
            }
        }
        // ---- epilogue: normalize, scatter to token-major bf16 buffer ----
#pragma unroll
        for (int j = 0; j < 8; ++j) {
            int qr = q0 + j + half * 8;
            if (qr < 392) {
                float inv = 1.0f / lrow[j];
                int tok = (dir == 0) ? (base + sn * 392 + qr)
                                     : (base + (qr % 56) * 56 + sn * 7 + qr / 56);
                Ob[(size_t)tok * CDIM + ho + nlo]      = f2bf(acc0[j] * inv);
                Ob[(size_t)tok * CDIM + ho + 16 + nlo] = f2bf(acc1[j] * inv);
            }
        }
    }
}

// ---------------------------------------------------------------------------
// Stage 4: proj GEMM  out[b,c,h,w] = attn[tok,:] . Wproj[c,:] + b_proj[c]
//   grid (196, 16), block 256; fp32 NCHW store to d_out.
// ---------------------------------------------------------------------------
__global__ __launch_bounds__(256) void gemm_proj_kernel(const u16* __restrict__ At,
                                                        const u16* __restrict__ W,
                                                        const float* __restrict__ bias,
                                                        float* __restrict__ out) {
    const int wave = threadIdx.x >> 5, lane = threadIdx.x & 31;
    const int nlo = lane & 15, half = lane >> 4;
    const int m0 = blockIdx.x * 128 + wave * 16;
    const int n0 = blockIdx.y * 16;

    v8f acc = {};
    const u16* arow = At + (size_t)(m0 + nlo) * CDIM;
    __builtin_prefetch(W + (size_t)(n0 + nlo) * CDIM, 0, 0);   // global_prefetch_b8
#pragma unroll
    for (int kk = 0; kk < 256; kk += 32) {
        v16bf a = load_afrag(arow + kk, CDIM, lane);
        v16bf b = load_bfrag_rowmajor(W, CDIM, n0, kk, lane);
        acc = __builtin_amdgcn_wmma_f32_16x16x32_bf16(false, a, false, b,
                                                      (short)0, acc, false, false);
    }
    const int o = n0 + nlo;
    const float bv = bias[o];
#pragma unroll
    for (int j = 0; j < 8; ++j) {
        int tok = m0 + j + half * 8;
        int b = tok / HWPIX;
        int p = tok - b * HWPIX;
        out[(size_t)b * (CDIM * HWPIX) + (size_t)o * HWPIX + p] = acc[j] + bv;
    }
}

// ---------------------------------------------------------------------------
// Host launcher
// ---------------------------------------------------------------------------
extern "C" void kernel_launch(void* const* d_in, const int* in_sizes, int n_in,
                              void* d_out, int out_size, void* d_ws, size_t ws_size,
                              hipStream_t stream) {
    (void)in_sizes; (void)n_in; (void)out_size; (void)ws_size;
    const float* x      = (const float*)d_in[0];   // [8,256,56,56]
    const float* w_qkv  = (const float*)d_in[1];   // [768,256]
    const float* b_qkv  = (const float*)d_in[2];   // [768]
    const float* w_dw   = (const float*)d_in[3];   // [256,1,3,3]
    const float* b_dw   = (const float*)d_in[4];   // [256]
    const float* w_proj = (const float*)d_in[5];   // [256,256]
    const float* b_proj = (const float*)d_in[6];   // [256]
    float* out = (float*)d_out;

    // workspace carve-up (all sizes multiples of 256B)
    char* ws = (char*)d_ws;
    float* Vf32   = (float*)ws;                 ws += (size_t)NTOK * CDIM * 4;  // 25.7 MB
    u16* Xt       = (u16*)ws;                   ws += (size_t)NTOK * CDIM * 2;
    u16* Qb       = (u16*)ws;                   ws += (size_t)NTOK * CDIM * 2;
    u16* Kb       = (u16*)ws;                   ws += (size_t)NTOK * CDIM * 2;
    u16* Vfin     = (u16*)ws;                   ws += (size_t)NTOK * CDIM * 2;
    u16* Attn     = (u16*)ws;                   ws += (size_t)NTOK * CDIM * 2;
    u16* Wqkv_bf  = (u16*)ws;                   ws += (size_t)768 * 256 * 2;
    u16* Wproj_bf = (u16*)ws;                   ws += (size_t)256 * 256 * 2;

    // 0) converts
    cvt_x_kernel<<<NTOK, 256, 0, stream>>>(x, Xt);
    cvt_w_kernel<<<768, 256, 0, stream>>>(w_qkv, Wqkv_bf, 768 * 256);
    cvt_w_kernel<<<256, 256, 0, stream>>>(w_proj, Wproj_bf, 256 * 256);
    // 1) qkv gemm (q scaled, v fp32)
    gemm_qkv_kernel<<<dim3(196, 48), 256, 0, stream>>>(Xt, Wqkv_bf, b_qkv, Qb, Kb, Vf32);
    // 2) depthwise residual on v
    dwconv_kernel<<<NTOK, 256, 0, stream>>>(Vf32, w_dw, b_dw, Vfin);
    // 3) stripe attention (512 stripes)
    stripe_attn_kernel<<<512, 256, 0, stream>>>(Qb, Kb, Vfin, Attn);
    // 4) projection gemm -> fp32 NCHW output
    gemm_proj_kernel<<<dim3(196, 16), 256, 0, stream>>>(Attn, Wproj_bf, b_proj, out);
}
